// PointNet2Seg_16381005267631
// MI455X (gfx1250) — matrix operand order repaired and checked by reading, not compiled, not run on hardware
//
#include <hip/hip_runtime.h>
#include <hip/hip_bf16.h>

// ---------------------------------------------------------------------------
// PointNet++ segmentation forward for MI455X (gfx1250).
// All pointwise MLPs (SA grouped MLPs, FP MLPs, seg head) are mapped onto
// V_WMMA_F32_16X16X32_F16 (f16 inputs, f32 accumulate) with a fused
// BN-affine + ReLU epilogue. SA modules fuse group-build -> 3-layer MLP ->
// channel max-pool inside one workgroup using dynamic LDS ping-pong buffers
// (<= 256 KB, within the 320 KB/WGP budget). Dense MLP layers stage their
// A-tiles into LDS with GLOBAL_LOAD_ASYNC_TO_LDS_B128 (ASYNCcnt) so all 8
// waves of a block feed WMMA from LDS instead of 8x-redundant global reads.
// FPS is a serial LDS-resident argmax loop (one workgroup per batch).
// Weights are converted to K-padded f16 once per call (deterministic).
// ---------------------------------------------------------------------------

typedef _Float16 half_t;
typedef __attribute__((ext_vector_type(16))) _Float16 v16h;
typedef __attribute__((ext_vector_type(8)))  _Float16 v8h;
typedef __attribute__((ext_vector_type(8)))  float    v8f;

#define KBN 0.9999950000374997f   // 1/sqrt(1+1e-5)

// ----------------------- gfx1250 async copy helpers ------------------------
// Per-lane 16B async copy global->LDS, tracked by ASYNCcnt. LDS operand is
// the 32-bit LDS byte offset (low 32 bits of a generic shared pointer per the
// shared aperture).
__device__ __forceinline__ void async_copy_b128(void* lds, const void* g) {
  unsigned loff = (unsigned)(unsigned long long)lds;
  unsigned long long ga = (unsigned long long)g;
  asm volatile("global_load_async_to_lds_b128 %0, %1, off"
               :: "v"(loff), "v"(ga) : "memory");
}
__device__ __forceinline__ void async_wait0() {
#if defined(__has_builtin)
#if __has_builtin(__builtin_amdgcn_s_wait_asynccnt)
  __builtin_amdgcn_s_wait_asynccnt(0);
  return;
#endif
#endif
  asm volatile("s_wait_asynccnt 0" ::: "memory");
}

// ---------------- WMMA fragment loaders (ISA 7.12.2 layouts) ---------------
// A (16x32 f16, MxK): lane L holds row L%16; lanes 0-15 K = {0..7,16..23},
// lanes 16-31 K = {8..15,24..31}.
__device__ __forceinline__ v16h load_a_frag(const half_t* buf, int stride,
                                            int lane, int kBase) {
  const half_t* p = buf + (size_t)(lane & 15) * stride + kBase + ((lane >> 4) << 3);
  v8h lo = *(const v8h*)(p);
  v8h hi = *(const v8h*)(p + 16);
  return __builtin_shufflevector(lo, hi, 0,1,2,3,4,5,6,7,8,9,10,11,12,13,14,15);
}

// B (32x16 f16, KxN), sourced from row-major W[c_out][Kpad] (B[k][n]=W[n][k]):
// lane L holds column L%16; lanes 0-15 K = {0..7,16..23}, lanes 16-31 K = {8..15,24..31}.
__device__ __forceinline__ v16h load_b_frag(const half_t* Wt, int Kp, int nBase,
                                            int lane, int kBase) {
  const half_t* p = Wt + (size_t)(nBase + (lane & 15)) * Kp + kBase + ((lane >> 4) << 3);
  v8h lo = *(const v8h*)(p);
  v8h hi = *(const v8h*)(p + 16);
  return __builtin_shufflevector(lo, hi, 0,1,2,3,4,5,6,7,8,9,10,11,12,13,14,15);
}

// One MLP layer on a 64-row LDS tile: out[64,cOut] = relu(in[64,Kp] x W^T *s + b)
// 4 x (cOut/16) WMMA tiles distributed over 8 wave32s. blockDim.x == 256.
__device__ __forceinline__ void gemm_layer(const half_t* in, int Kp,
                                           const half_t* Wt, const float* gg,
                                           const float* bb, int cOut,
                                           half_t* out, int stride, int tid) {
  int wave = tid >> 5, lane = tid & 31;
  int total = 4 * (cOut >> 4);
  for (int t = wave; t < total; t += 8) {
    int mT = t & 3, nT = t >> 2;
    v8f acc = {};
    const half_t* abase = in + (size_t)(mT * 16) * stride;
    for (int kt = 0; kt < Kp; kt += 32) {
      v16h a  = load_a_frag(abase, stride, lane, kt);
      v16h bf = load_b_frag(Wt, Kp, nT * 16, lane, kt);
      acc = __builtin_amdgcn_wmma_f32_16x16x32_f16(false, a, false, bf,
                                                   (short)0, acc, false, false);
    }
    int n  = nT * 16 + (lane & 15);
    int rh = lane >> 4;
    float sc = gg[n] * KBN, bi = bb[n];
#pragma unroll
    for (int r = 0; r < 8; ++r) {
      float v = fmaxf(acc[r] * sc + bi, 0.f);
      out[(size_t)(mT * 16 + rh * 8 + r) * stride + n] = (half_t)v;
    }
  }
}

// ------------------------------ utility kernels -----------------------------
__global__ void convert_w_kernel(const float* w, half_t* out, int cin, int Kpad) {
  int o = blockIdx.x;
  for (int c = threadIdx.x; c < Kpad; c += blockDim.x)
    out[(size_t)o * Kpad + c] = (half_t)(c < cin ? w[(size_t)o * cin + c] : 0.f);
}

__global__ void split_points_kernel(const float* pts, float* xyz, float* feat,
                                    int N, int total) {
  int t = blockIdx.x * blockDim.x + threadIdx.x;
  if (t >= total) return;
  int b = t / N, n = t - b * N;
  for (int c = 0; c < 3; ++c) {
    xyz [(size_t)t * 3 + c] = pts[((size_t)b * 6 + c)     * N + n];
    feat[(size_t)t * 3 + c] = pts[((size_t)b * 6 + 3 + c) * N + n];
  }
}

// Farthest point sampling: one workgroup per batch; mind[] in dynamic LDS.
__global__ void fps_kernel(const float* xyz, int N, int m, int* idx) {
  extern __shared__ char sm[];
  float* mind = (float*)sm;
  float* redv = (float*)(sm + (size_t)N * 4);
  int*   redi = (int*)(redv + 256);
  int*   bc   = redi + 256;
  int b = blockIdx.x, tid = threadIdx.x;
  const float* X = xyz + (size_t)b * N * 3;
  for (int n = tid; n < N; n += blockDim.x) mind[n] = 1e10f;
  if (tid == 0) idx[(size_t)b * m] = 0;
  __syncthreads();
  int last = 0;
  for (int it = 1; it < m; ++it) {
    float lx = X[last * 3], ly = X[last * 3 + 1], lz = X[last * 3 + 2];
    float bv = -1.f; int bi = 0;
    for (int n = tid; n < N; n += blockDim.x) {
      float dx = X[n * 3] - lx, dy = X[n * 3 + 1] - ly, dz = X[n * 3 + 2] - lz;
      float d = dx * dx + dy * dy + dz * dz;
      float mm = mind[n];
      if (d < mm) { mm = d; mind[n] = mm; }
      if (mm > bv) { bv = mm; bi = n; }
    }
    redv[tid] = bv; redi[tid] = bi;
    __syncthreads();
    for (int s = 128; s > 0; s >>= 1) {
      if (tid < s) {
        float v2 = redv[tid + s]; int i2 = redi[tid + s];
        if (v2 > redv[tid] || (v2 == redv[tid] && i2 < redi[tid])) {
          redv[tid] = v2; redi[tid] = i2;
        }
      }
      __syncthreads();
    }
    if (tid == 0) { bc[0] = redi[0]; idx[(size_t)b * m + it] = redi[0]; }
    __syncthreads();
    last = bc[0];
    __syncthreads();
  }
}

__global__ void gather_xyz_kernel(const float* xyz, const int* idx, float* out,
                                  int Nin, int m, int total) {
  int t = blockIdx.x * blockDim.x + threadIdx.x;
  if (t >= total) return;
  int b = t / m;
  int j = idx[t];
  for (int c = 0; c < 3; ++c)
    out[(size_t)t * 3 + c] = xyz[((size_t)b * Nin + j) * 3 + c];
}

// First-64-in-radius-by-index (matches reference score/top_k semantics).
__global__ void ball_query_kernel(const float* xyz, const float* nxyz, int N,
                                  int m, float r2, int* out, int total) {
  int t = blockIdx.x * blockDim.x + threadIdx.x;
  if (t >= total) return;
  int b = t / m;
  const float* X = xyz + (size_t)b * N * 3;
  float cx = nxyz[t * 3], cy = nxyz[t * 3 + 1], cz = nxyz[t * 3 + 2];
  int* o = out + (size_t)t * 64;
  int cnt = 0;
  for (int j = 0; j < N && cnt < 64; ++j) {
    float dx = X[j * 3] - cx, dy = X[j * 3 + 1] - cy, dz = X[j * 3 + 2] - cz;
    if (dx * dx + dy * dy + dz * dz <= r2) o[cnt++] = j;
  }
  if (cnt == 0) { for (int k2 = 0; k2 < 64; ++k2) o[k2] = N - 1; }
  else { int f = o[0]; for (int k2 = cnt; k2 < 64; ++k2) o[k2] = f; }
}

// Fused SA module: group-build -> 3x (WMMA MLP layer) -> channel max-pool.
// blockDim.x = 256 (8 wave32s). Dynamic LDS: 2 * 64 * stride * sizeof(f16).
__global__ void sa_fused_kernel(const float* xyz, const float* feat,
                                const float* nxyz, const int* ballIdx,
                                const half_t* W1, const float* g1, const float* b1,
                                const half_t* W2, const float* g2, const float* b2,
                                const half_t* W3, const float* g3, const float* b3,
                                float* fout, int Nin, int m, int Cf, int Kpad0,
                                int c1, int c2, int c3, int stride) {
  extern __shared__ char smemRaw[];
  half_t* bufA = (half_t*)smemRaw;
  half_t* bufB = bufA + (size_t)64 * stride;
  int b = blockIdx.y, i = blockIdx.x, tid = threadIdx.x;
  int cin = 3 + Cf;
  const int* nb = ballIdx + ((size_t)b * m + i) * 64;
  float cx = nxyz[((size_t)b * m + i) * 3 + 0];
  float cy = nxyz[((size_t)b * m + i) * 3 + 1];
  float cz = nxyz[((size_t)b * m + i) * 3 + 2];
  for (int e = tid; e < 64 * Kpad0; e += blockDim.x) {
    int r = e / Kpad0, c = e - r * Kpad0;
    int j = nb[r];
    float v;
    if (c == 0)       v = xyz[((size_t)b * Nin + j) * 3 + 0] - cx;
    else if (c == 1)  v = xyz[((size_t)b * Nin + j) * 3 + 1] - cy;
    else if (c == 2)  v = xyz[((size_t)b * Nin + j) * 3 + 2] - cz;
    else if (c < cin) v = feat[((size_t)b * Nin + j) * Cf + (c - 3)];
    else              v = 0.f;
    bufA[(size_t)r * stride + c] = (half_t)v;
  }
  __syncthreads();
  gemm_layer(bufA, Kpad0, W1, g1, b1, c1, bufB, stride, tid);
  __syncthreads();
  gemm_layer(bufB, c1, W2, g2, b2, c2, bufA, stride, tid);
  __syncthreads();
  gemm_layer(bufA, c2, W3, g3, b3, c3, bufB, stride, tid);
  __syncthreads();
  for (int c = tid; c < c3; c += blockDim.x) {
    float mx = -1e30f;
    for (int r = 0; r < 64; ++r)
      mx = fmaxf(mx, (float)bufB[(size_t)r * stride + c]);
    fout[((size_t)b * m + i) * c3 + c] = mx;
  }
}

// Dense MLP layer: each wave owns one 16x16 output tile; the block's shared
// 16-row A-tile is staged global->LDS with async b128 copies (ASYNCcnt).
// grid = (rows/16, cOut/128), blockDim.x = 256, dyn LDS = 16*Kp*2 bytes.
__global__ void mlp_dense_kernel(const half_t* act, int Kp, const half_t* Wt,
                                 const float* gg, const float* bb, int cOut,
                                 half_t* out_h, float* out_f) {
  extern __shared__ char smemRaw[];
  half_t* atile = (half_t*)smemRaw;                 // [16][Kp]
  int tid = threadIdx.x;
  int mBase = blockIdx.x * 16;
  const half_t* gsrc = act + (size_t)mBase * Kp;    // contiguous 16*Kp halves
  int chunks = (16 * Kp) >> 3;                      // 16-byte chunks
  for (int e = tid; e < chunks; e += blockDim.x)
    async_copy_b128(atile + (size_t)e * 8, gsrc + (size_t)e * 8);
  async_wait0();
  __syncthreads();

  int wave = tid >> 5, lane = tid & 31;
  int nBase = (blockIdx.y * 8 + wave) * 16;
  v8f acc = {};
  for (int kt = 0; kt < Kp; kt += 32) {
    v16h a  = load_a_frag(atile, Kp, lane, kt);
    v16h bf = load_b_frag(Wt, Kp, nBase, lane, kt);
    acc = __builtin_amdgcn_wmma_f32_16x16x32_f16(false, a, false, bf,
                                                 (short)0, acc, false, false);
  }
  int n  = nBase + (lane & 15);
  int rh = lane >> 4;
  float sc = gg[n] * KBN, bi = bb[n];
#pragma unroll
  for (int r = 0; r < 8; ++r) {
    int row = mBase + rh * 8 + r;
    float v = fmaxf(acc[r] * sc + bi, 0.f);
    if (out_h) out_h[(size_t)row * cOut + n] = (half_t)v;
    if (out_f) out_f[(size_t)row * cOut + n] = v;
  }
}

// 3-NN (smallest d2, ties -> lower index) + normalized inverse-distance weights.
__global__ void knn3_kernel(const float* dxyz, const float* sxyz, int Nd, int Ns,
                            int* idx3, float* w3, int total) {
  int t = blockIdx.x * blockDim.x + threadIdx.x;
  if (t >= total) return;
  int b = t / Nd;
  float px = dxyz[t * 3], py = dxyz[t * 3 + 1], pz = dxyz[t * 3 + 2];
  const float* S = sxyz + (size_t)b * Ns * 3;
  float d0 = 1e30f, d1 = 1e30f, d2 = 1e30f;
  int i0 = 0, i1 = 0, i2 = 0;
  for (int j = 0; j < Ns; ++j) {
    float dx = S[j * 3] - px, dy = S[j * 3 + 1] - py, dz = S[j * 3 + 2] - pz;
    float d = dx * dx + dy * dy + dz * dz;
    if (d < d0)      { d2 = d1; i2 = i1; d1 = d0; i1 = i0; d0 = d; i0 = j; }
    else if (d < d1) { d2 = d1; i2 = i1; d1 = d; i1 = j; }
    else if (d < d2) { d2 = d; i2 = j; }
  }
  float w0 = 1.f / (d0 + 1e-8f), w1 = 1.f / (d1 + 1e-8f), w2 = 1.f / (d2 + 1e-8f);
  float s = w0 + w1 + w2;
  idx3[t * 3] = i0; idx3[t * 3 + 1] = i1; idx3[t * 3 + 2] = i2;
  w3[t * 3] = w0 / s; w3[t * 3 + 1] = w1 / s; w3[t * 3 + 2] = w2 / s;
}

// Build FP input rows: [interp(sparse_feat) | dense_feat | zero-pad] as f16.
__global__ void interp_concat_kernel(const float* sfeat, const float* dfeat,
                                     const int* idx3, const float* w3, int Nd,
                                     int Ns, int Cs, int Cd, int Kpad,
                                     half_t* out) {
  int p = blockIdx.x;            // b*Nd + n
  int b = p / Nd;
  const int* id = idx3 + (size_t)p * 3;
  const float* w = w3 + (size_t)p * 3;
  const float* s0 = sfeat + ((size_t)b * Ns + id[0]) * Cs;
  const float* s1 = sfeat + ((size_t)b * Ns + id[1]) * Cs;
  const float* s2 = sfeat + ((size_t)b * Ns + id[2]) * Cs;
  float w0 = w[0], w1 = w[1], w2 = w[2];
  for (int c = threadIdx.x; c < Kpad; c += blockDim.x) {
    float v;
    if (c < Cs)           v = w0 * s0[c] + w1 * s1[c] + w2 * s2[c];
    else if (c < Cs + Cd) v = dfeat[(size_t)p * Cd + (c - Cs)];
    else                  v = 0.f;
    out[(size_t)p * Kpad + c] = (half_t)v;
  }
}

__global__ void score_kernel(const half_t* x, const float* sw, const float* sb,
                             const float* sg, const float* sbeta, float* out,
                             int total) {
  int p = blockIdx.x * blockDim.x + threadIdx.x;
  if (p >= total) return;
  float s = 0.f;
  for (int c = 0; c < 128; ++c) s += (float)x[(size_t)p * 128 + c] * sw[c];
  s += sb[0];
  s = s * (sg[0] * KBN) + sbeta[0];
  out[p] = 1.f / (1.f + __expf(-s));
}

__global__ void transpose_out_kernel(const float* sf, float* out, int N, int C,
                                     int total) {
  int t = blockIdx.x * blockDim.x + threadIdx.x;
  if (t >= total) return;
  int b = t / (N * C);
  int rem = t - b * N * C;
  int n = rem / C, c = rem - n * C;
  out[((size_t)b * C + c) * N + n] = sf[t];
}

// ------------------------------- host driver --------------------------------
extern "C" void kernel_launch(void* const* d_in, const int* in_sizes, int n_in,
                              void* d_out, int out_size, void* d_ws, size_t ws_size,
                              hipStream_t stream) {
  const int B = 2, N0 = 12800, N1 = 5120, N2 = 1024, N3 = 256;
  const float R1 = 0.02f, R2 = 0.08f, R3 = 0.32f;
  const float* points = (const float*)d_in[0];
  auto P = [&](int i) { return (const float*)d_in[i]; };

  char* base = (char*)d_ws;
  size_t off = 0;
  auto alloc = [&](size_t bytes) -> void* {
    off = (off + 255) & ~(size_t)255;
    void* p = base + off;
    off += bytes;
    return p;
  };

  // Layer table: {param index of w (g=+1, b=+2), c_in, c_out}. Param order:
  // points, sa[0..2] x 3 layers x (w,g,b), fp[0..2], seg, score_w/b/g/beta.
  struct LD { int idx, cin, cout; };
  const LD L[20] = {
      {1, 6, 128},   {4, 128, 128},  {7, 128, 256},
      {10, 259, 256},{13, 256, 256}, {16, 256, 512},
      {19, 515, 512},{22, 512, 512}, {25, 512, 1024},
      {28, 1536, 1024},{31, 1024, 1024},
      {34, 1280, 512},{37, 512, 512},
      {40, 515, 256},{43, 256, 256},{46, 256, 256},
      {49, 256, 512},{52, 512, 256},{55, 256, 256},{58, 256, 128}};
  half_t* wh[20];
  int kp[20];
  for (int l = 0; l < 20; ++l) {
    kp[l] = ((L[l].cin + 31) / 32) * 32;
    wh[l] = (half_t*)alloc((size_t)L[l].cout * kp[l] * sizeof(half_t));
    convert_w_kernel<<<dim3(L[l].cout), dim3(256), 0, stream>>>(
        P(L[l].idx), wh[l], L[l].cin, kp[l]);
  }

  float* xyz0  = (float*)alloc((size_t)B * N0 * 3 * 4);
  float* feat0 = (float*)alloc((size_t)B * N0 * 3 * 4);
  float* xyz1  = (float*)alloc((size_t)B * N1 * 3 * 4);
  float* xyz2  = (float*)alloc((size_t)B * N2 * 3 * 4);
  float* xyz3  = (float*)alloc((size_t)B * N3 * 3 * 4);
  float* feat1 = (float*)alloc((size_t)B * N1 * 256 * 4);
  float* feat2 = (float*)alloc((size_t)B * N2 * 512 * 4);
  float* feat3 = (float*)alloc((size_t)B * N3 * 1024 * 4);
  int* fi1 = (int*)alloc((size_t)B * N1 * 4);
  int* fi2 = (int*)alloc((size_t)B * N2 * 4);
  int* fi3 = (int*)alloc((size_t)B * N3 * 4);
  int* bq1 = (int*)alloc((size_t)B * N1 * 64 * 4);
  int* bq2 = (int*)alloc((size_t)B * N2 * 64 * 4);
  int* bq3 = (int*)alloc((size_t)B * N3 * 64 * 4);
  int*   ki1 = (int*)  alloc((size_t)B * N2 * 3 * 4);
  float* kw1 = (float*)alloc((size_t)B * N2 * 3 * 4);
  int*   ki2 = (int*)  alloc((size_t)B * N1 * 3 * 4);
  float* kw2 = (float*)alloc((size_t)B * N1 * 3 * 4);
  int*   ki3 = (int*)  alloc((size_t)B * N0 * 3 * 4);
  float* kw3 = (float*)alloc((size_t)B * N0 * 3 * 4);
  const size_t ACT_ELEMS = (size_t)B * N0 * 544;   // largest rows*Kpad
  half_t* actA = (half_t*)alloc(ACT_ELEMS * sizeof(half_t));
  half_t* actB = (half_t*)alloc(ACT_ELEMS * sizeof(half_t));
  float* fp1o = (float*)alloc((size_t)B * N2 * 1024 * 4);
  float* fp2o = (float*)alloc((size_t)B * N1 * 512 * 4);
  float* sfo  = (float*)alloc((size_t)B * N0 * 256 * 4);

  // level-0 split
  split_points_kernel<<<(B * N0 + 255) / 256, 256, 0, stream>>>(
      points, xyz0, feat0, N0, B * N0);

  // -------- SA1: 12800 -> 5120, c 6 -> (128,128,256), r=0.02 --------
  fps_kernel<<<B, 256, (size_t)N0 * 4 + 2064, stream>>>(xyz0, N0, N1, fi1);
  gather_xyz_kernel<<<(B * N1 + 255) / 256, 256, 0, stream>>>(xyz0, fi1, xyz1, N0, N1, B * N1);
  ball_query_kernel<<<(B * N1 + 255) / 256, 256, 0, stream>>>(xyz0, xyz1, N0, N1, R1 * R1, bq1, B * N1);
  sa_fused_kernel<<<dim3(N1, B), 256, (size_t)2 * 64 * 256 * 2, stream>>>(
      xyz0, feat0, xyz1, bq1,
      wh[0], P(2), P(3), wh[1], P(5), P(6), wh[2], P(8), P(9),
      feat1, N0, N1, 3, kp[0], 128, 128, 256, 256);

  // -------- SA2: 5120 -> 1024, c 259 -> (256,256,512), r=0.08 --------
  fps_kernel<<<B, 256, (size_t)N1 * 4 + 2064, stream>>>(xyz1, N1, N2, fi2);
  gather_xyz_kernel<<<(B * N2 + 255) / 256, 256, 0, stream>>>(xyz1, fi2, xyz2, N1, N2, B * N2);
  ball_query_kernel<<<(B * N2 + 255) / 256, 256, 0, stream>>>(xyz1, xyz2, N1, N2, R2 * R2, bq2, B * N2);
  sa_fused_kernel<<<dim3(N2, B), 256, (size_t)2 * 64 * 512 * 2, stream>>>(
      xyz1, feat1, xyz2, bq2,
      wh[3], P(11), P(12), wh[4], P(14), P(15), wh[5], P(17), P(18),
      feat2, N1, N2, 256, kp[3], 256, 256, 512, 512);

  // -------- SA3: 1024 -> 256, c 515 -> (512,512,1024), r=0.32 --------
  fps_kernel<<<B, 256, (size_t)N2 * 4 + 2064, stream>>>(xyz2, N2, N3, fi3);
  gather_xyz_kernel<<<(B * N3 + 255) / 256, 256, 0, stream>>>(xyz2, fi3, xyz3, N2, N3, B * N3);
  ball_query_kernel<<<(B * N3 + 255) / 256, 256, 0, stream>>>(xyz2, xyz3, N2, N3, R3 * R3, bq3, B * N3);
  sa_fused_kernel<<<dim3(N3, B), 256, (size_t)2 * 64 * 1024 * 2, stream>>>(
      xyz2, feat2, xyz3, bq3,
      wh[6], P(20), P(21), wh[7], P(23), P(24), wh[8], P(26), P(27),
      feat3, N2, N3, 512, kp[6], 512, 512, 1024, 1024);

  // -------- FP1: sparse L3 -> dense L2, c_in 1536 -> (1024,1024) --------
  knn3_kernel<<<(B * N2 + 255) / 256, 256, 0, stream>>>(xyz2, xyz3, N2, N3, ki1, kw1, B * N2);
  interp_concat_kernel<<<B * N2, 256, 0, stream>>>(feat3, feat2, ki1, kw1, N2, N3, 1024, 512, 1536, actA);
  mlp_dense_kernel<<<dim3((B * N2) / 16, 8), 256, (size_t)16 * 1536 * 2, stream>>>(actA, 1536, wh[9],  P(29), P(30), 1024, actB, nullptr);
  mlp_dense_kernel<<<dim3((B * N2) / 16, 8), 256, (size_t)16 * 1024 * 2, stream>>>(actB, 1024, wh[10], P(32), P(33), 1024, nullptr, fp1o);

  // -------- FP2: sparse L2 -> dense L1, c_in 1280 -> (512,512) --------
  knn3_kernel<<<(B * N1 + 255) / 256, 256, 0, stream>>>(xyz1, xyz2, N1, N2, ki2, kw2, B * N1);
  interp_concat_kernel<<<B * N1, 256, 0, stream>>>(fp1o, feat1, ki2, kw2, N1, N2, 1024, 256, 1280, actA);
  mlp_dense_kernel<<<dim3((B * N1) / 16, 4), 256, (size_t)16 * 1280 * 2, stream>>>(actA, 1280, wh[11], P(35), P(36), 512, actB, nullptr);
  mlp_dense_kernel<<<dim3((B * N1) / 16, 4), 256, (size_t)16 * 512 * 2,  stream>>>(actB, 512,  wh[12], P(38), P(39), 512, nullptr, fp2o);

  // -------- FP3: sparse L1 -> dense L0, c_in 515 -> (256,256,256) --------
  knn3_kernel<<<(B * N0 + 255) / 256, 256, 0, stream>>>(xyz0, xyz1, N0, N1, ki3, kw3, B * N0);
  interp_concat_kernel<<<B * N0, 256, 0, stream>>>(fp2o, feat0, ki3, kw3, N0, N1, 512, 3, 544, actA);
  mlp_dense_kernel<<<dim3((B * N0) / 16, 2), 256, (size_t)16 * 544 * 2, stream>>>(actA, 544, wh[13], P(41), P(42), 256, actB, nullptr);
  mlp_dense_kernel<<<dim3((B * N0) / 16, 2), 256, (size_t)16 * 256 * 2, stream>>>(actB, 256, wh[14], P(44), P(45), 256, actA, nullptr);
  mlp_dense_kernel<<<dim3((B * N0) / 16, 2), 256, (size_t)16 * 256 * 2, stream>>>(actA, 256, wh[15], P(47), P(48), 256, actB, sfo);

  // -------- seg head: 256 -> 512 -> 256 -> 256 -> 128 --------
  mlp_dense_kernel<<<dim3((B * N0) / 16, 4), 256, (size_t)16 * 256 * 2, stream>>>(actB, 256, wh[16], P(50), P(51), 512, actA, nullptr);
  mlp_dense_kernel<<<dim3((B * N0) / 16, 2), 256, (size_t)16 * 512 * 2, stream>>>(actA, 512, wh[17], P(53), P(54), 256, actB, nullptr);
  mlp_dense_kernel<<<dim3((B * N0) / 16, 2), 256, (size_t)16 * 256 * 2, stream>>>(actB, 256, wh[18], P(56), P(57), 256, actA, nullptr);
  mlp_dense_kernel<<<dim3((B * N0) / 16, 1), 256, (size_t)16 * 256 * 2, stream>>>(actA, 256, wh[19], P(59), P(60), 128, actB, nullptr);

  // -------- outputs --------
  float* out_feat  = (float*)d_out;                          // [B,256,N0]
  float* out_score = out_feat + (size_t)B * 256 * N0;        // [B,N0]
  score_kernel<<<(B * N0 + 255) / 256, 256, 0, stream>>>(
      actB, P(61), P(62), P(63), P(64), out_score, B * N0);
  transpose_out_kernel<<<(B * N0 * 256 + 255) / 256, 256, 0, stream>>>(
      sfo, out_feat, N0, 256, B * N0 * 256);
}
